// CovarianceMatrix_51350628991246
// MI455X (gfx1250) — matrix operand order repaired and checked by reading, compiled, not run on hardware
//
#include <hip/hip_runtime.h>
#include <hip/hip_bf16.h>
#include <math.h>
#include <stdint.h>

// ---------------------------------------------------------------------------
// CovarianceMatrix on MI455X (gfx1250):
//   rho = (x * cov_kernel) @ x^T  via bf16x3-split WMMA (v_wmma_f32_16x16x32_bf16)
//   Fast path: pre-split x into bf16 hi/lo in workspace, then GEMM stages tiles
//   with GLOBAL_LOAD_ASYNC_TO_LDS_B128 (ASYNCcnt), double-buffered.
//   diag overwritten with softplus(x @ var_kernel + var_bias) + 1e-8
// ---------------------------------------------------------------------------

#define MAT_N 8192
#define MAT_D 512

typedef __attribute__((ext_vector_type(16))) __bf16 bf16x16;
typedef __attribute__((ext_vector_type(8)))  __bf16 bf16x8;
typedef __attribute__((ext_vector_type(8)))  float  v8f;

static constexpr int BM   = 128;   // block tile M
static constexpr int BN   = 128;   // block tile N
static constexpr int BK   = 32;    // k-step (== WMMA K)
static constexpr int LDSK = 40;    // padded LDS row stride (bf16 elems) -> no bank conflicts
static constexpr int TILE_ELEMS = BM * LDSK;        // per LDS tile, bf16 elems
static constexpr int TILE_BYTES = TILE_ELEMS * 2;   // 10240 B

__device__ __forceinline__ bf16x16 join8(bf16x8 a, bf16x8 b) {
    return __builtin_shufflevector(a, b, 0,1,2,3,4,5,6,7,8,9,10,11,12,13,14,15);
}

// 16B global -> LDS async DMA (gfx1250), tracked by ASYNCcnt
__device__ __forceinline__ void async_cp16(unsigned lds_off, const void* g) {
    asm volatile("global_load_async_to_lds_b128 %0, %1, off"
                 :: "v"(lds_off), "v"((unsigned long long)(uintptr_t)g)
                 : "memory");
}
__device__ __forceinline__ void wait_async_le8() {
    asm volatile("s_wait_asynccnt 0x8" ::: "memory");
}
__device__ __forceinline__ void wait_async_0() {
    asm volatile("s_wait_asynccnt 0x0" ::: "memory");
}

// ---------------------------------------------------------------------------
// Pre-pass: split x (and x*cov) into bf16 hi/lo arrays (row-major [8192][512]).
// Pure bandwidth: reads 16MB, writes 32MB.
// ---------------------------------------------------------------------------
__global__ __launch_bounds__(256)
void cov_presplit(const float* __restrict__ x, const float* __restrict__ cov,
                  __bf16* __restrict__ aHi, __bf16* __restrict__ aLo,
                  __bf16* __restrict__ bHi, __bf16* __restrict__ bLo)
{
    const size_t base = ((size_t)blockIdx.x * 256 + threadIdx.x) * 8;
    const int k = (int)(base & (MAT_D - 1));

    const float4* src = (const float4*)(x + base);
    float4 t0 = src[0], t1 = src[1];
    float v[8] = { t0.x, t0.y, t0.z, t0.w, t1.x, t1.y, t1.z, t1.w };

    bf16x8 ah, al, bh, bl;
#pragma unroll
    for (int i = 0; i < 8; ++i) {
        float bv = v[i];
        __bf16 g = (__bf16)bv;
        bh[i] = g;
        bl[i] = (__bf16)(bv - (float)g);
        float av = bv * cov[k + i];
        __bf16 h = (__bf16)av;
        ah[i] = h;
        al[i] = (__bf16)(av - (float)h);
    }
    *(bf16x8*)(aHi + base) = ah;
    *(bf16x8*)(aLo + base) = al;
    *(bf16x8*)(bHi + base) = bh;
    *(bf16x8*)(bLo + base) = bl;
}

// ---------------------------------------------------------------------------
// Fast GEMM: pre-split bf16 operands, async-to-LDS double-buffered staging.
// ---------------------------------------------------------------------------
__device__ __forceinline__ void issue_stage(
    const __bf16* __restrict__ aHi, const __bf16* __restrict__ aLo,
    const __bf16* __restrict__ bHi, const __bf16* __restrict__ bLo,
    __bf16* sbuf, int i0, int j0, int k0, int tid)
{
    const int row   = tid >> 2;                 // 0..63
    const int chunk = tid & 3;                  // 0..3 (16B chunks of a 64B row)
    const size_t aoff = ((size_t)(i0 + row) * MAT_D + k0 + chunk * 8);
    const size_t boff = ((size_t)(j0 + row) * MAT_D + k0 + chunk * 8);
    const size_t rstep = (size_t)64 * MAT_D;    // +64 rows

    const unsigned l0 = (unsigned)(uintptr_t)sbuf + row * (LDSK * 2) + chunk * 16;
    const unsigned l1 = l0 + 64 * (LDSK * 2);

    async_cp16(l0,                  aHi + aoff);
    async_cp16(l0 + TILE_BYTES,     aLo + aoff);
    async_cp16(l0 + 2 * TILE_BYTES, bHi + boff);
    async_cp16(l0 + 3 * TILE_BYTES, bLo + boff);
    async_cp16(l1,                  aHi + aoff + rstep);
    async_cp16(l1 + TILE_BYTES,     aLo + aoff + rstep);
    async_cp16(l1 + 2 * TILE_BYTES, bHi + boff + rstep);
    async_cp16(l1 + 3 * TILE_BYTES, bLo + boff + rstep);
}

__global__ __launch_bounds__(256)
void cov_gram_wmma_async(const __bf16* __restrict__ aHi, const __bf16* __restrict__ aLo,
                         const __bf16* __restrict__ bHi, const __bf16* __restrict__ bLo,
                         float* __restrict__ out)
{
    __shared__ __bf16 smem[2][4][TILE_ELEMS];   // [buf][Ahi,Alo,Bhi,Blo][tile]

    const int tid = threadIdx.x;
    const int i0  = blockIdx.y * BM;
    const int j0  = blockIdx.x * BN;

    const int wave = tid >> 5;
    const int lane = tid & 31;
    const int wm   = wave >> 2;         // 0..1 : M offset wm*64
    const int wn   = wave & 3;          // 0..3 : N offset wn*32
    const int lrow = lane & 15;
    const int lhi  = lane >> 4;

    const v8f vzero = {0.f,0.f,0.f,0.f,0.f,0.f,0.f,0.f};
    v8f acc[4][2];
#pragma unroll
    for (int mt = 0; mt < 4; ++mt)
#pragma unroll
        for (int nt = 0; nt < 2; ++nt) acc[mt][nt] = vzero;

    constexpr int NSTAGE = MAT_D / BK;  // 16

    issue_stage(aHi, aLo, bHi, bLo, &smem[0][0][0], i0, j0, 0, tid);

    for (int s = 0; s < NSTAGE; ++s) {
        if (s + 1 < NSTAGE) {
            issue_stage(aHi, aLo, bHi, bLo, &smem[(s + 1) & 1][0][0],
                        i0, j0, (s + 1) * BK, tid);
            wait_async_le8();   // stage s landed (async loads complete in order)
        } else {
            wait_async_0();
        }
        __syncthreads();        // all waves' stage-s data visible

        const __bf16* sA_hi = &smem[s & 1][0][0];
        const __bf16* sA_lo = &smem[s & 1][1][0];
        const __bf16* sB_hi = &smem[s & 1][2][0];
        const __bf16* sB_lo = &smem[s & 1][3][0];

        // A (16x32 bf16): lane<16 row=lrow holds K {0..7,16..23}; lane>=16 K {8..15,24..31}
        bf16x16 aH[4], aL[4], bH[2], bL[2];
#pragma unroll
        for (int mt = 0; mt < 4; ++mt) {
            const int r = (wm * 64 + mt * 16 + lrow) * LDSK;
            aH[mt] = join8(*(const bf16x8*)&sA_hi[r + lhi * 8],
                           *(const bf16x8*)&sA_hi[r + 16 + lhi * 8]);
            aL[mt] = join8(*(const bf16x8*)&sA_lo[r + lhi * 8],
                           *(const bf16x8*)&sA_lo[r + 16 + lhi * 8]);
        }
        // B (32x16 bf16): lane<16 col=lrow holds K {0..7,16..23}; lane>=16 K {8..15,24..31}
#pragma unroll
        for (int nt = 0; nt < 2; ++nt) {
            const int r = (wn * 32 + nt * 16 + lrow) * LDSK + lhi * 8;
            bH[nt] = join8(*(const bf16x8*)&sB_hi[r], *(const bf16x8*)&sB_hi[r + 16]);
            bL[nt] = join8(*(const bf16x8*)&sB_lo[r], *(const bf16x8*)&sB_lo[r + 16]);
        }

#pragma unroll
        for (int mt = 0; mt < 4; ++mt) {
#pragma unroll
            for (int nt = 0; nt < 2; ++nt) {
                acc[mt][nt] = __builtin_amdgcn_wmma_f32_16x16x32_bf16(
                    false, aH[mt], false, bH[nt], (short)0, acc[mt][nt], false, false);
                acc[mt][nt] = __builtin_amdgcn_wmma_f32_16x16x32_bf16(
                    false, aH[mt], false, bL[nt], (short)0, acc[mt][nt], false, false);
                acc[mt][nt] = __builtin_amdgcn_wmma_f32_16x16x32_bf16(
                    false, aL[mt], false, bH[nt], (short)0, acc[mt][nt], false, false);
            }
        }
        __syncthreads();        // readers done before buffer (s&1) is overwritten
    }

    // epilogue: VGPR v holds row (v + lhi*8), col lrow; NT stores (268MB stream > L2)
#pragma unroll
    for (int mt = 0; mt < 4; ++mt) {
#pragma unroll
        for (int nt = 0; nt < 2; ++nt) {
            const int gm = i0 + wm * 64 + mt * 16 + lhi * 8;
            const int gn = j0 + wn * 32 + nt * 16 + lrow;
#pragma unroll
            for (int v = 0; v < 8; ++v) {
                __builtin_nontemporal_store(acc[mt][nt][v],
                                            &out[(size_t)(gm + v) * MAT_N + gn]);
            }
        }
    }
}

// ---------------------------------------------------------------------------
// Fallback GEMM (no workspace): fused fp32->bf16 hi/lo conversion in-kernel.
// ---------------------------------------------------------------------------
__global__ __launch_bounds__(256)
void cov_gram_wmma_fused(const float* __restrict__ x,
                         const float* __restrict__ cov,
                         float* __restrict__ out)
{
    __shared__ __bf16 sA_hi[TILE_ELEMS];
    __shared__ __bf16 sA_lo[TILE_ELEMS];
    __shared__ __bf16 sB_hi[TILE_ELEMS];
    __shared__ __bf16 sB_lo[TILE_ELEMS];
    __shared__ float  sCov[MAT_D];

    const int tid = threadIdx.x;
    const int i0  = blockIdx.y * BM;
    const int j0  = blockIdx.x * BN;

    for (int k = tid; k < MAT_D; k += 256) sCov[k] = cov[k];

    const int srow = tid >> 1;
    const int shalf = tid & 1;

    const int wave = tid >> 5;
    const int lane = tid & 31;
    const int wm   = wave >> 2;
    const int wn   = wave & 3;
    const int lrow = lane & 15;
    const int lhi  = lane >> 4;

    const v8f vzero = {0.f,0.f,0.f,0.f,0.f,0.f,0.f,0.f};
    v8f acc[4][2];
#pragma unroll
    for (int mt = 0; mt < 4; ++mt)
#pragma unroll
        for (int nt = 0; nt < 2; ++nt) acc[mt][nt] = vzero;

    for (int k0 = 0; k0 < MAT_D; k0 += BK) {
        __syncthreads();
        {
            const float4* aSrc = (const float4*)(x + (size_t)(i0 + srow) * MAT_D + k0 + shalf * 16);
            const float4* bSrc = (const float4*)(x + (size_t)(j0 + srow) * MAT_D + k0 + shalf * 16);
            float a[16], b[16];
#pragma unroll
            for (int q = 0; q < 4; ++q) {
                float4 t = aSrc[q];
                a[4*q+0] = t.x; a[4*q+1] = t.y; a[4*q+2] = t.z; a[4*q+3] = t.w;
            }
#pragma unroll
            for (int q = 0; q < 4; ++q) {
                float4 t = bSrc[q];
                b[4*q+0] = t.x; b[4*q+1] = t.y; b[4*q+2] = t.z; b[4*q+3] = t.w;
            }
            bf16x8 ah[2], al[2], bh[2], bl[2];
#pragma unroll
            for (int c = 0; c < 2; ++c) {
#pragma unroll
                for (int i = 0; i < 8; ++i) {
                    const int e = c * 8 + i;
                    float av = a[e] * sCov[k0 + shalf * 16 + e];
                    __bf16 h = (__bf16)av;
                    ah[c][i] = h;
                    al[c][i] = (__bf16)(av - (float)h);
                    float bv = b[e];
                    __bf16 g = (__bf16)bv;
                    bh[c][i] = g;
                    bl[c][i] = (__bf16)(bv - (float)g);
                }
            }
            const int base = srow * LDSK + shalf * 16;
            *(bf16x8*)&sA_hi[base]     = ah[0];
            *(bf16x8*)&sA_hi[base + 8] = ah[1];
            *(bf16x8*)&sA_lo[base]     = al[0];
            *(bf16x8*)&sA_lo[base + 8] = al[1];
            *(bf16x8*)&sB_hi[base]     = bh[0];
            *(bf16x8*)&sB_hi[base + 8] = bh[1];
            *(bf16x8*)&sB_lo[base]     = bl[0];
            *(bf16x8*)&sB_lo[base + 8] = bl[1];
        }
        if (k0 + BK < MAT_D) {
            __builtin_prefetch(x + (size_t)(i0 + srow) * MAT_D + k0 + BK + shalf * 16, 0, 0);
            __builtin_prefetch(x + (size_t)(j0 + srow) * MAT_D + k0 + BK + shalf * 16, 0, 0);
        }
        __syncthreads();

        bf16x16 aH[4], aL[4], bH[2], bL[2];
#pragma unroll
        for (int mt = 0; mt < 4; ++mt) {
            const int r = (wm * 64 + mt * 16 + lrow) * LDSK;
            aH[mt] = join8(*(const bf16x8*)&sA_hi[r + lhi * 8],
                           *(const bf16x8*)&sA_hi[r + 16 + lhi * 8]);
            aL[mt] = join8(*(const bf16x8*)&sA_lo[r + lhi * 8],
                           *(const bf16x8*)&sA_lo[r + 16 + lhi * 8]);
        }
#pragma unroll
        for (int nt = 0; nt < 2; ++nt) {
            const int r = (wn * 32 + nt * 16 + lrow) * LDSK + lhi * 8;
            bH[nt] = join8(*(const bf16x8*)&sB_hi[r], *(const bf16x8*)&sB_hi[r + 16]);
            bL[nt] = join8(*(const bf16x8*)&sB_lo[r], *(const bf16x8*)&sB_lo[r + 16]);
        }

#pragma unroll
        for (int mt = 0; mt < 4; ++mt) {
#pragma unroll
            for (int nt = 0; nt < 2; ++nt) {
                acc[mt][nt] = __builtin_amdgcn_wmma_f32_16x16x32_bf16(
                    false, aH[mt], false, bH[nt], (short)0, acc[mt][nt], false, false);
                acc[mt][nt] = __builtin_amdgcn_wmma_f32_16x16x32_bf16(
                    false, aH[mt], false, bL[nt], (short)0, acc[mt][nt], false, false);
                acc[mt][nt] = __builtin_amdgcn_wmma_f32_16x16x32_bf16(
                    false, aL[mt], false, bH[nt], (short)0, acc[mt][nt], false, false);
            }
        }
    }

#pragma unroll
    for (int mt = 0; mt < 4; ++mt) {
#pragma unroll
        for (int nt = 0; nt < 2; ++nt) {
            const int gm = i0 + wm * 64 + mt * 16 + lhi * 8;
            const int gn = j0 + wn * 32 + nt * 16 + lrow;
#pragma unroll
            for (int v = 0; v < 8; ++v) {
                __builtin_nontemporal_store(acc[mt][nt][v],
                                            &out[(size_t)(gm + v) * MAT_N + gn]);
            }
        }
    }
}

// ---------------------------------------------------------------------------
// Diagonal: out[i,i] = softplus(x[i,:] @ var_kernel + var_bias) + 1e-8
// ---------------------------------------------------------------------------
__global__ __launch_bounds__(256)
void cov_diag_softplus(const float* __restrict__ x,
                       const float* __restrict__ vk,
                       const float* __restrict__ vb,
                       float* __restrict__ out)
{
    const int wave = blockIdx.x * 8 + (threadIdx.x >> 5);
    const int lane = threadIdx.x & 31;
    if (wave >= MAT_N) return;

    const float* xr = x + (size_t)wave * MAT_D;
    float s = 0.f;
#pragma unroll
    for (int t = 0; t < MAT_D / 32; ++t) {
        const int k = t * 32 + lane;
        s = fmaf(xr[k], vk[k], s);
    }
#pragma unroll
    for (int off = 16; off > 0; off >>= 1)
        s += __shfl_down(s, off, 32);

    if (lane == 0) {
        float z = s + vb[0];
        float sp = (z > 20.f) ? z : log1pf(expf(z));
        out[(size_t)wave * MAT_N + wave] = sp + 1e-8f;
    }
}

// ---------------------------------------------------------------------------
extern "C" void kernel_launch(void* const* d_in, const int* in_sizes, int n_in,
                              void* d_out, int out_size, void* d_ws, size_t ws_size,
                              hipStream_t stream) {
    const float* x   = (const float*)d_in[0];   // [8192, 512]
    const float* cov = (const float*)d_in[1];   // [512]
    const float* vk  = (const float*)d_in[2];   // [512, 1]
    const float* vb  = (const float*)d_in[3];   // [1]
    float* out = (float*)d_out;                 // [8192, 8192]

    dim3 grid(MAT_N / BN, MAT_N / BM);          // 64 x 64
    const size_t arr = (size_t)MAT_N * MAT_D;   // elems per split array
    const size_t need = 4 * arr * sizeof(__bf16);   // 32 MB

    if (ws_size >= need) {
        __bf16* aHi = (__bf16*)d_ws;
        __bf16* aLo = aHi + arr;
        __bf16* bHi = aLo + arr;
        __bf16* bLo = bHi + arr;
        cov_presplit<<<(MAT_N * MAT_D / 8) / 256, 256, 0, stream>>>(x, cov, aHi, aLo, bHi, bLo);
        cov_gram_wmma_async<<<grid, 256, 0, stream>>>(aHi, aLo, bHi, bLo, out);
    } else {
        cov_gram_wmma_fused<<<grid, 256, 0, stream>>>(x, cov, out);
    }

    cov_diag_softplus<<<MAT_N / 8, 256, 0, stream>>>(x, vk, vb, out);
}
